// KhmerOCR_91027536871640
// MI455X (gfx1250) — compile-verified, matrix operand order checked by reading, not run on hardware
//
#include <hip/hip_runtime.h>

// ============================================================================
// KhmerOCR forward for MI455X (gfx1250, wave32).
//  - All GEMMs via v_wmma_f32_16x16x32_f16 (f32 accumulate).
//  - Weights + activations kept in f16 for GEMM operands; BOTH tiles of every
//    dense GEMM stream through the Tensor Data Mover into double-buffered LDS
//    (s_wait_tensorcnt(2) => chunk i+1 DMA overlaps chunk i WMMAs).
//  - Implicit-GEMM conv: weight tile via TDM, im2col gather via VALU (overlap).
//  - Macro-tile 32(M) x 64(N), 4 waves/block, 2 accumulators/wave.
// ============================================================================

typedef __attribute__((ext_vector_type(16))) _Float16 v16h;
typedef __attribute__((ext_vector_type(8)))  float    v8f;
typedef __attribute__((ext_vector_type(4)))  _Float16 h4;
typedef __attribute__((ext_vector_type(4)))  unsigned int u32x4;
typedef __attribute__((ext_vector_type(8)))  int       i32x8;
typedef __attribute__((ext_vector_type(4)))  int       i32x4;

#define KCH 32      // WMMA K chunk
#define LDT 40      // padded LDS leading dim (halves): 64B row + 16B pad

// ---- WMMA fragment helpers -------------------------------------------------
__device__ __forceinline__ v16h frag_ld(const _Float16* t) {
  int lane = threadIdx.x & 31;
  int r  = lane & 15;
  int kb = (lane & 16) ? 8 : 0;
  v16h f;
#pragma unroll
  for (int i = 0; i < 8; ++i) {
    f[i]     = t[r * LDT + kb + i];
    f[i + 8] = t[r * LDT + kb + 16 + i];
  }
  return f;
}

__device__ __forceinline__ v8f wmma16(v16h a, v16h b, v8f c) {
  return __builtin_amdgcn_wmma_f32_16x16x32_f16(false, a, false, b,
                                                (short)0, c, false, false);
}

__device__ __forceinline__ void frag_st(float* C, int ldc,
                                        _Float16* Ch, int ldch,
                                        int m0, int n0, int M, int N, v8f acc,
                                        const float* bias, int act) {
  int lane = threadIdx.x & 31;
  int col  = lane & 15;
  int rb   = (lane & 16) ? 8 : 0;
  int n = n0 + col;
  if (n >= N) return;
  float bv = bias ? bias[n] : 0.f;
#pragma unroll
  for (int i = 0; i < 8; ++i) {
    int m = m0 + rb + i;
    if (m < M) {
      float v = acc[i] + bv;
      if (act == 1) v = tanhf(v);
      C[(size_t)m * ldc + n] = v;
      if (Ch) Ch[(size_t)m * ldch + n] = (_Float16)v;
    }
  }
}

// ---- Tensor Data Mover: 2-D f16 tile -> padded LDS rows (stride LDT) ------
// D# per CDNA5 ISA ch.8. pad_interval=3 (16 DW = 64B row), pad_amount=1
// (2 DW = 16B) => LDS row stride = LDT halves. tensor dims => HW OOB zero.
__device__ __forceinline__ void tdm_tile_f16(unsigned lds_bytes,
                                             const _Float16* gsrc,
                                             int rem_k, int rem_n,
                                             int stride_elems,
                                             int tile_k, int tile_n) {
  unsigned long long ga = (unsigned long long)(size_t)gsrc;
  u32x4 g0;
  g0[0] = 1u;                                         // count=1, user mode
  g0[1] = lds_bytes;                                  // lds_addr
  g0[2] = (unsigned)(ga & 0xffffffffu);               // global_addr lo
  g0[3] = (unsigned)((ga >> 32) & 0x01ffffffu) | (2u << 30); // hi | type=2
  if (rem_k < 0) rem_k = 0;
  if (rem_n < 0) rem_n = 0;
  i32x8 g1;
  g1[0] = (1 << 16) | (1 << 20) | (3 << 22) | (1 << 25); // 2B, pad 2DW/16DW
  g1[1] = (rem_k & 0xffff) << 16;
  g1[2] = ((rem_k >> 16) & 0xffff) | ((rem_n & 0xffff) << 16);
  g1[3] = ((rem_n >> 16) & 0xffff) | ((tile_k & 0xffff) << 16);
  g1[4] = tile_n & 0xffff;
  g1[5] = stride_elems;
  g1[6] = 0;
  g1[7] = 0;
  i32x4 z4 = {0, 0, 0, 0};
  i32x8 z8 = {0, 0, 0, 0, 0, 0, 0, 0};
  __builtin_amdgcn_tensor_load_to_lds(g0, g1, z4, z4, z8, 0);
}

// ---- GEMM: C[M,N] = Ah[M,K](f16) @ Wh[N,K](f16).T  (+bias, opt tanh) ------
// Requires M%32==0, K%32==0 (true for all call sites). Both tiles via TDM,
// double-buffered; s_wait_tensorcnt(2) pipelines DMA under WMMA.
__global__ void gemm_wmma(const _Float16* __restrict__ Ah, int lda,
                          const _Float16* __restrict__ Wh, int ldb,
                          float* __restrict__ C, int ldc,
                          _Float16* __restrict__ Ch, int ldch,
                          const float* __restrict__ bias, int act,
                          int M, int N, int K) {
  __shared__ _Float16 As[2][32 * LDT];
  __shared__ _Float16 Bs[2][64 * LDT];
  int m0 = blockIdx.y * 32;
  int n0 = blockIdx.x * 64;
  int tid = threadIdx.x, wave = tid >> 5;
  v8f acc0 = {}, acc1 = {};
  int nk = K / KCH;
  if (tid == 0) {   // prime buffer 0
    tdm_tile_f16((unsigned)(size_t)&As[0][0], Ah + (size_t)m0 * lda,
                 K, M - m0, lda, KCH, 32);
    tdm_tile_f16((unsigned)(size_t)&Bs[0][0], Wh + (size_t)n0 * ldb,
                 K, N - n0, ldb, KCH, 64);
  }
  for (int ic = 0; ic < nk; ++ic) {
    int ib = ic & 1;
    if (tid == 0) {
      if (ic + 1 < nk) {  // prefetch next chunk, wait only for current
        int kn = (ic + 1) * KCH;
        tdm_tile_f16((unsigned)(size_t)&As[ib ^ 1][0],
                     Ah + (size_t)m0 * lda + kn, K - kn, M - m0, lda, KCH, 32);
        tdm_tile_f16((unsigned)(size_t)&Bs[ib ^ 1][0],
                     Wh + (size_t)n0 * ldb + kn, K - kn, N - n0, ldb, KCH, 64);
        __builtin_amdgcn_s_wait_tensorcnt(2);
      } else {
        __builtin_amdgcn_s_wait_tensorcnt(0);
      }
    }
    __syncthreads();
    v16h a0 = frag_ld(&As[ib][0]);
    v16h a1 = frag_ld(&As[ib][16 * LDT]);
    v16h bf = frag_ld(&Bs[ib][(wave * 16) * LDT]);
    acc0 = wmma16(a0, bf, acc0);
    acc1 = wmma16(a1, bf, acc1);
    __syncthreads();
  }
  frag_st(C, ldc, Ch, ldch, m0,      n0 + wave * 16, M, N, acc0, bias, act);
  frag_st(C, ldc, Ch, ldch, m0 + 16, n0 + wave * 16, M, N, acc1, bias, act);
}

// ---- Implicit-GEMM conv: M=Cout, N=B*Ho*Wo, K=Cin*KH*KW -------------------
// Weight tile (f16) via TDM; im2col tile gathered/converted by VALU (overlap).
__global__ void conv_wmma(const float* __restrict__ x,
                          const _Float16* __restrict__ wh,
                          float* __restrict__ y,
                          int Bn, int Cin, int H, int W,
                          int Cout, int KH, int KW, int stride, int pad,
                          int Ho, int Wo) {
  __shared__ _Float16 As[32 * LDT];
  __shared__ _Float16 Bs[64 * LDT];
  int K = Cin * KH * KW;
  int N = Bn * Ho * Wo;
  int m0 = blockIdx.y * 32;
  int n0 = blockIdx.x * 64;
  int tid = threadIdx.x, wave = tid >> 5;
  v8f acc0 = {}, acc1 = {};
  for (int kc = 0; kc < K; kc += KCH) {
    if (tid == 0)   // DMA 32x32 weight tile while VALU gathers im2col
      tdm_tile_f16((unsigned)(size_t)&As[0],
                   wh + (size_t)m0 * K + kc, K - kc, Cout - m0, K, KCH, 32);
    for (int u = tid; u < 64 * 8; u += 128) {
      int rr = u >> 3, kg = (u & 7) * 4;
      int n = n0 + rr;
      h4 hv = {};
      if (n < N) {
        int wo = n % Wo; int t = n / Wo; int ho = t % Ho; int b = t / Ho;
        int hbase = ho * stride - pad;
        int wbase = wo * stride - pad;
#pragma unroll
        for (int j = 0; j < 4; ++j) {
          int k = kc + kg + j;
          float v = 0.f;
          if (k < K) {
            int kw = k % KW; int t2 = k / KW; int kh = t2 % KH; int ci = t2 / KH;
            int hi = hbase + kh, wi = wbase + kw;
            if (hi >= 0 && hi < H && wi >= 0 && wi < W)
              v = x[(((size_t)b * Cin + ci) * H + hi) * W + wi];
          }
          hv[j] = (_Float16)v;
        }
      }
      *(h4*)&Bs[rr * LDT + kg] = hv;
    }
    if (tid == 0) __builtin_amdgcn_s_wait_tensorcnt(0);
    __syncthreads();
    v16h a0 = frag_ld(As);
    v16h a1 = frag_ld(&As[16 * LDT]);
    v16h bf = frag_ld(&Bs[(wave * 16) * LDT]);
    acc0 = wmma16(a0, bf, acc0);
    acc1 = wmma16(a1, bf, acc1);
    __syncthreads();
  }
  int lane = tid & 31;
  int col  = lane & 15;
  int rb   = (lane & 16) ? 8 : 0;
  int n = n0 + wave * 16 + col;
  if (n < N) {
    int wo = n % Wo; int t = n / Wo; int ho = t % Ho; int b = t / Ho;
#pragma unroll
    for (int i = 0; i < 8; ++i) {
      int m = m0 + rb + i;
      if (m < Cout)
        y[(((size_t)b * Cout + m) * Ho + ho) * Wo + wo] = acc0[i];
      int m2 = m0 + 16 + rb + i;
      if (m2 < Cout)
        y[(((size_t)b * Cout + m2) * Ho + ho) * Wo + wo] = acc1[i];
    }
  }
}

// ---- fp32 -> f16 ----------------------------------------------------------
__global__ void cvt_to_h(const float* __restrict__ s, _Float16* __restrict__ d,
                         int n) {
  int i = blockIdx.x * blockDim.x + threadIdx.x;
  if (i < n) d[i] = (_Float16)s[i];
}

// ---- BatchNorm (training-mode batch stats) --------------------------------
__global__ void bn_stats(const float* __restrict__ x, float* __restrict__ st,
                         int Bn, int C, int HW) {
  int c = blockIdx.x;
  __shared__ float ssum[256], ssq[256];
  float s = 0.f, q = 0.f;
  int cnt = Bn * HW;
  for (int i = threadIdx.x; i < cnt; i += 256) {
    int b = i / HW, r = i % HW;
    float v = x[((size_t)b * C + c) * HW + r];
    s += v; q += v * v;
  }
  ssum[threadIdx.x] = s; ssq[threadIdx.x] = q;
  __syncthreads();
  for (int off = 128; off > 0; off >>= 1) {
    if (threadIdx.x < off) {
      ssum[threadIdx.x] += ssum[threadIdx.x + off];
      ssq[threadIdx.x]  += ssq[threadIdx.x + off];
    }
    __syncthreads();
  }
  if (threadIdx.x == 0) {
    float mean = ssum[0] / cnt;
    float var  = ssq[0] / cnt - mean * mean;
    st[c]     = mean;
    st[C + c] = rsqrtf(var + 1e-5f);
  }
}

__global__ void bn_apply(const float* __restrict__ x, const float* __restrict__ st,
                         const float* __restrict__ g, const float* __restrict__ bt,
                         const float* __restrict__ x2, const float* __restrict__ st2,
                         const float* __restrict__ g2, const float* __restrict__ b2,
                         float* __restrict__ y, int C, int HW, size_t total, int relu) {
  size_t i = (size_t)blockIdx.x * blockDim.x + threadIdx.x;
  if (i >= total) return;
  int c = (int)((i / HW) % C);
  float v = g[c] * (x[i] - st[c]) * st[C + c] + bt[c];
  if (x2) v += g2[c] * (x2[i] - st2[c]) * st2[C + c] + b2[c];
  if (relu) v = fmaxf(v, 0.f);
  y[i] = v;
}

// ---- misc ------------------------------------------------------------------
__global__ void fill_zero(float* p, size_t n) {
  size_t i = (size_t)blockIdx.x * blockDim.x + threadIdx.x;
  if (i < n) p[i] = 0.f;
}
__global__ void fill_zero_h(_Float16* p, size_t n) {
  size_t i = (size_t)blockIdx.x * blockDim.x + threadIdx.x;
  if (i < n) p[i] = (_Float16)0.f;
}

// x4:[B,512,4,64] -> xs_h:[S=64,B,512] f16 (GEMM A operand)
__global__ void meanpool(const float* __restrict__ x, _Float16* __restrict__ xsh) {
  int i = blockIdx.x * blockDim.x + threadIdx.x;
  if (i >= 64 * 64 * 512) return;
  int c = i % 512; int t = i / 512; int b = t % 64; int s = t / 64;
  float a = 0.f;
#pragma unroll
  for (int h = 0; h < 4; ++h) a += x[(((size_t)b * 512 + c) * 4 + h) * 64 + s];
  xsh[i] = (_Float16)(a * 0.25f);
}

// torch GRU gate order r,z,n; dual-writes h (f32 for elementwise consumers,
// f16 for the next step's WMMA A operand)
__global__ void gru_cell(const float* __restrict__ gi, const float* __restrict__ gh,
                         const float* __restrict__ hp, float* __restrict__ hn,
                         _Float16* __restrict__ hnh, int Bn, int Hd) {
  int i = blockIdx.x * blockDim.x + threadIdx.x;
  if (i >= Bn * Hd) return;
  int b = i / Hd, j = i % Hd;
  const float* gib = gi + (size_t)b * 3 * Hd;
  const float* ghb = gh + (size_t)b * 3 * Hd;
  float r = 1.f / (1.f + expf(-(gib[j]          + ghb[j])));
  float z = 1.f / (1.f + expf(-(gib[Hd + j]     + ghb[Hd + j])));
  float n = tanhf(gib[2 * Hd + j] + r * ghb[2 * Hd + j]);
  float h = (1.f - z) * n + z * hp[i];
  hn[i] = h;
  hnh[i] = (_Float16)h;
}

__global__ void assemble_encout(const float* __restrict__ yf, const float* __restrict__ yb,
                                float* __restrict__ eo, _Float16* __restrict__ eoh) {
  int i = blockIdx.x * blockDim.x + threadIdx.x;
  if (i >= 64 * 64 * 512) return;
  int c = i % 512; int t = i / 512; int s = t % 64; int b = t / 64;
  float v = (c < 256) ? yf[((size_t)s * 64 + b) * 256 + c]
                      : yb[((size_t)(63 - s) * 64 + b) * 256 + (c - 256)];
  eo[i] = v;
  eoh[i] = (_Float16)v;
}

__global__ void concat2h(const float* __restrict__ a, int na,
                         const float* __restrict__ b, int nb,
                         _Float16* __restrict__ o, int Bn) {
  int i = blockIdx.x * blockDim.x + threadIdx.x;
  int w = na + nb;
  if (i >= Bn * w) return;
  int bb = i / w, j = i % w;
  o[i] = (_Float16)((j < na) ? a[(size_t)bb * na + j] : b[(size_t)bb * nb + (j - na)]);
}

__global__ void attn_kernel(const float* __restrict__ s1, const float* __restrict__ ep,
                            const float* __restrict__ vv, const float* __restrict__ eo,
                            float* __restrict__ wgt) {
  int b = blockIdx.x, tid = threadIdx.x;
  __shared__ float sc[64];
  __shared__ float red[256];
  __shared__ float mx, sm;
  int s = tid >> 2, sub = tid & 3;
  float part = 0.f;
  for (int j = sub; j < 256; j += 4) {
    float e = tanhf(s1[(size_t)b * 256 + j] + ep[((size_t)b * 64 + s) * 256 + j]);
    part += e * vv[j];
  }
  red[tid] = part;
  __syncthreads();
  if (sub == 0) sc[s] = red[tid] + red[tid + 1] + red[tid + 2] + red[tid + 3];
  __syncthreads();
  if (tid == 0) {
    float m = sc[0];
    for (int k = 1; k < 64; ++k) m = fmaxf(m, sc[k]);
    mx = m;
  }
  __syncthreads();
  if (tid < 64) sc[tid] = expf(sc[tid] - mx);
  __syncthreads();
  if (tid == 0) {
    float t = 0.f;
    for (int k = 0; k < 64; ++k) t += sc[k];
    sm = t;
  }
  __syncthreads();
  if (tid < 64) sc[tid] /= sm;
  __syncthreads();
  for (int d = tid; d < 512; d += 256) {
    float a = 0.f;
    for (int k = 0; k < 64; ++k) a += sc[k] * eo[((size_t)b * 64 + k) * 512 + d];
    wgt[(size_t)b * 512 + d] = a;
  }
}

// xcat(f16) = [emb(trg[:,t]) | weighted]
__global__ void build_xcat(const int* __restrict__ trg, int t,
                           const float* __restrict__ emb, const float* __restrict__ wgt,
                           _Float16* __restrict__ xc) {
  int i = blockIdx.x * blockDim.x + threadIdx.x;
  if (i >= 64 * 640) return;
  int b = i / 640, j = i % 640;
  float v = (j < 128) ? emb[(size_t)trg[b * 32 + t] * 128 + j]
                      : wgt[(size_t)b * 512 + (j - 128)];
  xc[i] = (_Float16)v;
}

// ocat(f16) = [h2 | weighted | emb]
__global__ void build_ocat(const float* __restrict__ h2, const float* __restrict__ wgt,
                           const int* __restrict__ trg, int t,
                           const float* __restrict__ emb, _Float16* __restrict__ oc) {
  int i = blockIdx.x * blockDim.x + threadIdx.x;
  if (i >= 64 * 896) return;
  int b = i / 896, j = i % 896;
  float v;
  if (j < 256)      v = h2[(size_t)b * 256 + j];
  else if (j < 768) v = wgt[(size_t)b * 512 + (j - 256)];
  else              v = emb[(size_t)trg[b * 32 + t] * 128 + (j - 768)];
  oc[i] = (_Float16)v;
}

// ============================================================================
static inline int cdiv(int a, int b) { return (a + b - 1) / b; }

extern "C" void kernel_launch(void* const* d_in, const int* in_sizes, int n_in,
                              void* d_out, int out_size, void* d_ws, size_t ws_size,
                              hipStream_t stream) {
  (void)in_sizes; (void)n_in; (void)out_size; (void)ws_size;
  const float* src = (const float*)d_in[0];
  const int*   trg = (const int*)d_in[1];
  auto F = [&](int i) { return (const float*)d_in[i]; };
  int p = 2;
  const float *Bc1[4], *Bg1[4], *Bb1[4], *Bc2[4], *Bg2[4], *Bb2[4], *Bsc[4], *Bsg[4], *Bsb[4];
  for (int k = 0; k < 4; ++k) {
    Bc1[k] = F(p++); Bg1[k] = F(p++); Bb1[k] = F(p++);
    Bc2[k] = F(p++); Bg2[k] = F(p++); Bb2[k] = F(p++);
    Bsc[k] = F(p++); Bsg[k] = F(p++); Bsb[k] = F(p++);
  }
  const float *WihF = F(p++), *WhhF = F(p++), *bihF = F(p++), *bhhF = F(p++);
  const float *WihB = F(p++), *WhhB = F(p++), *bihB = F(p++), *bhhB = F(p++);
  const float *Wfc = F(p++), *bfc = F(p++);
  const float *emb = F(p++), *Wa = F(p++), *ba = F(p++), *vv = F(p++);
  const float *WihD = F(p++), *WhhD = F(p++), *bihD = F(p++), *bhhD = F(p++);
  const float *Wo = F(p++), *bo = F(p++);

  float* ws = (float*)d_ws;
  float* dout = (float*)d_out;

  // big activation buffers (64 Mi floats each)
  float* X  = ws;
  float* T1 = ws + ((size_t)1 << 26);
  float* T2 = ws + ((size_t)2 << 26);
  float* SC = ws + ((size_t)3 << 26);
  float* ST1 = ws + ((size_t)4 << 26);
  float* ST2 = ST1 + 2048;
  float* ST3 = ST1 + 4096;
  size_t o = ((size_t)4 << 26) + 8192;
  float* GIF = ws + o; o += (size_t)64 * 64 * 768;
  float* GIB = ws + o; o += (size_t)64 * 64 * 768;
  float* YF  = ws + o; o += (size_t)64 * 64 * 256;
  float* YB  = ws + o; o += (size_t)64 * 64 * 256;
  float* GHB = ws + o; o += (size_t)64 * 768;
  float* HF0 = ws + o; o += (size_t)64 * 256;
  float* HB0 = ws + o; o += (size_t)64 * 256;
  float* HD0 = ws + o; o += (size_t)64 * 256;
  float* HD1 = ws + o; o += (size_t)64 * 256;
  float* EO  = ws + o; o += (size_t)64 * 64 * 512;
  float* EPRJ= ws + o; o += (size_t)64 * 64 * 256;
  float* S1b = ws + o; o += (size_t)64 * 256;
  float* WGT = ws + o; o += (size_t)64 * 512;
  float* GID = ws + o; o += (size_t)64 * 768;
  float* GHD = ws + o; o += (size_t)64 * 768;

  // f16 pool: converted weights + f16 activations (GEMM A operands)
  _Float16* whp = (_Float16*)(ws + o);
  size_t wo_ = 0;
  auto CV = [&](const float* sp, size_t n) {
    _Float16* d = whp + wo_;
    wo_ += (n + 7) & ~(size_t)7;
    cvt_to_h<<<cdiv((int)n, 256), 256, 0, stream>>>(sp, d, (int)n);
    return (const _Float16*)d;
  };
  auto AH = [&](size_t n) {
    _Float16* d = whp + wo_;
    wo_ += (n + 7) & ~(size_t)7;
    return d;
  };

  const int Cin[4]  = {1, 64, 128, 256};
  const int Cout[4] = {64, 128, 256, 512};
  const int Hs[5] = {32, 32, 16, 8, 4};
  const int Ws_[5] = {512, 512, 256, 128, 64};
  const int strd[4] = {1, 2, 2, 2};

  const _Float16 *Hc1[4], *Hc2[4], *Hsc[4];
  for (int k = 0; k < 4; ++k) {
    Hc1[k] = CV(Bc1[k], (size_t)Cout[k] * Cin[k] * 9);
    Hc2[k] = CV(Bc2[k], (size_t)Cout[k] * Cout[k] * 9);
    Hsc[k] = CV(Bsc[k], (size_t)Cout[k] * Cin[k]);
  }
  const _Float16* HihF = CV(WihF, 768 * 512);
  const _Float16* HhhF = CV(WhhF, 768 * 256);
  const _Float16* HihB = CV(WihB, 768 * 512);
  const _Float16* HhhB = CV(WhhB, 768 * 256);
  const _Float16* HWfc = CV(Wfc, 256 * 512);
  const _Float16* HWa  = CV(Wa, 256 * 768);
  const _Float16* HihD = CV(WihD, 768 * 640);
  const _Float16* HhhD = CV(WhhD, 768 * 256);
  const _Float16* HWo  = CV(Wo, 140 * 896);

  _Float16* XSH  = AH((size_t)64 * 64 * 512);
  _Float16* YFH  = AH((size_t)64 * 64 * 256);
  _Float16* YBH  = AH((size_t)64 * 64 * 256);
  _Float16* HF0H = AH((size_t)64 * 256);
  _Float16* HB0H = AH((size_t)64 * 256);
  _Float16* HD0H = AH((size_t)64 * 256);
  _Float16* HD1H = AH((size_t)64 * 256);
  _Float16* EOH  = AH((size_t)64 * 64 * 512);
  _Float16* HCATH= AH((size_t)64 * 512);
  _Float16* XCH  = AH((size_t)64 * 640);
  _Float16* OCH  = AH((size_t)64 * 896);

  // ---------------- CNN backbone ----------------
  const float* xcur = src;
  for (int k = 0; k < 4; ++k) {
    int H = Hs[k], W = Ws_[k], Ho = Hs[k + 1], Wo2 = Ws_[k + 1];
    int ci = Cin[k], co = Cout[k], st = strd[k];
    int N = 64 * Ho * Wo2;
    int HW = Ho * Wo2;
    size_t osz = (size_t)64 * co * HW;
    dim3 cgrid(cdiv(N, 64), cdiv(co, 32));
    conv_wmma<<<cgrid, 128, 0, stream>>>(xcur, Hc1[k], T1, 64, ci, H, W, co, 3, 3, st, 1, Ho, Wo2);
    bn_stats<<<co, 256, 0, stream>>>(T1, ST1, 64, co, HW);
    bn_apply<<<cdiv((int)osz, 256), 256, 0, stream>>>(T1, ST1, Bg1[k], Bb1[k],
        nullptr, nullptr, nullptr, nullptr, T1, co, HW, osz, 1);
    conv_wmma<<<cgrid, 128, 0, stream>>>(T1, Hc2[k], T2, 64, co, Ho, Wo2, co, 3, 3, 1, 1, Ho, Wo2);
    bn_stats<<<co, 256, 0, stream>>>(T2, ST2, 64, co, HW);
    conv_wmma<<<cgrid, 128, 0, stream>>>(xcur, Hsc[k], SC, 64, ci, H, W, co, 1, 1, st, 0, Ho, Wo2);
    bn_stats<<<co, 256, 0, stream>>>(SC, ST3, 64, co, HW);
    bn_apply<<<cdiv((int)osz, 256), 256, 0, stream>>>(T2, ST2, Bg2[k], Bb2[k],
        SC, ST3, Bsg[k], Bsb[k], X, co, HW, osz, 1);
    xcur = X;
  }

  // ---------------- seq features + encoder GRU ----------------
  meanpool<<<cdiv(64 * 64 * 512, 256), 256, 0, stream>>>(X, XSH);
  {
    dim3 g(cdiv(768, 64), cdiv(4096, 32));
    gemm_wmma<<<g, 128, 0, stream>>>(XSH, 512, HihF, 512, GIF, 768, nullptr, 0,
                                     bihF, 0, 4096, 768, 512);
    gemm_wmma<<<g, 128, 0, stream>>>(XSH, 512, HihB, 512, GIB, 768, nullptr, 0,
                                     bihB, 0, 4096, 768, 512);
  }
  fill_zero<<<cdiv(64 * 256, 256), 256, 0, stream>>>(HF0, 64 * 256);
  fill_zero<<<cdiv(64 * 256, 256), 256, 0, stream>>>(HB0, 64 * 256);
  fill_zero_h<<<cdiv(64 * 256, 256), 256, 0, stream>>>(HF0H, 64 * 256);
  fill_zero_h<<<cdiv(64 * 256, 256), 256, 0, stream>>>(HB0H, 64 * 256);
  {
    dim3 gh(cdiv(768, 64), cdiv(64, 32));
    for (int t = 0; t < 64; ++t) {
      const float*    hpf  = t ? (YF  + (size_t)(t - 1) * 64 * 256) : HF0;
      const _Float16* hpfh = t ? (YFH + (size_t)(t - 1) * 64 * 256) : HF0H;
      gemm_wmma<<<gh, 128, 0, stream>>>(hpfh, 256, HhhF, 256, GHB, 768, nullptr, 0,
                                        bhhF, 0, 64, 768, 256);
      gru_cell<<<cdiv(64 * 256, 256), 256, 0, stream>>>(GIF + (size_t)t * 64 * 768, GHB, hpf,
          YF + (size_t)t * 64 * 256, YFH + (size_t)t * 64 * 256, 64, 256);
      const float*    hpb  = t ? (YB  + (size_t)(t - 1) * 64 * 256) : HB0;
      const _Float16* hpbh = t ? (YBH + (size_t)(t - 1) * 64 * 256) : HB0H;
      gemm_wmma<<<gh, 128, 0, stream>>>(hpbh, 256, HhhB, 256, GHB, 768, nullptr, 0,
                                        bhhB, 0, 64, 768, 256);
      gru_cell<<<cdiv(64 * 256, 256), 256, 0, stream>>>(GIB + (size_t)(63 - t) * 64 * 768, GHB, hpb,
          YB + (size_t)t * 64 * 256, YBH + (size_t)t * 64 * 256, 64, 256);
    }
  }
  assemble_encout<<<cdiv(64 * 64 * 512, 256), 256, 0, stream>>>(YF, YB, EO, EOH);
  concat2h<<<cdiv(64 * 512, 256), 256, 0, stream>>>(YF + (size_t)63 * 64 * 256, 256,
                                                    YB + (size_t)63 * 64 * 256, 256, HCATH, 64);
  {
    dim3 g(cdiv(256, 64), cdiv(64, 32));
    gemm_wmma<<<g, 128, 0, stream>>>(HCATH, 512, HWfc, 512, HD0, 256, HD0H, 256,
                                     bfc, 1, 64, 256, 512);   // tanh, dual out
  }
  {
    dim3 g(cdiv(256, 64), cdiv(4096, 32));
    gemm_wmma<<<g, 128, 0, stream>>>(EOH, 512, HWa + 256, 768, EPRJ, 256, nullptr, 0,
                                     ba, 0, 4096, 256, 512);
  }

  // ---------------- attention decoder ----------------
  float* hcur = HD0;  _Float16* hcurH = HD0H;
  float* hnxt = HD1;  _Float16* hnxtH = HD1H;
  dim3 gs1(cdiv(256, 64), cdiv(64, 32));
  dim3 ggate(cdiv(768, 64), cdiv(64, 32));
  dim3 gpred(cdiv(140, 64), cdiv(64, 32));
  for (int t = 0; t < 31; ++t) {
    gemm_wmma<<<gs1, 128, 0, stream>>>(hcurH, 256, HWa, 768, S1b, 256, nullptr, 0,
                                       nullptr, 0, 64, 256, 256);
    attn_kernel<<<64, 256, 0, stream>>>(S1b, EPRJ, vv, EO, WGT);
    build_xcat<<<cdiv(64 * 640, 256), 256, 0, stream>>>(trg, t, emb, WGT, XCH);
    gemm_wmma<<<ggate, 128, 0, stream>>>(XCH, 640, HihD, 640, GID, 768, nullptr, 0,
                                         bihD, 0, 64, 768, 640);
    gemm_wmma<<<ggate, 128, 0, stream>>>(hcurH, 256, HhhD, 256, GHD, 768, nullptr, 0,
                                         bhhD, 0, 64, 768, 256);
    gru_cell<<<cdiv(64 * 256, 256), 256, 0, stream>>>(GID, GHD, hcur, hnxt, hnxtH, 64, 256);
    build_ocat<<<cdiv(64 * 896, 256), 256, 0, stream>>>(hnxt, WGT, trg, t, emb, OCH);
    gemm_wmma<<<gpred, 128, 0, stream>>>(OCH, 896, HWo, 896,
                                         dout + (size_t)t * 140, 31 * 140, nullptr, 0,
                                         bo, 0, 64, 140, 896);
    float* tf = hcur; hcur = hnxt; hnxt = tf;
    _Float16* th = hcurH; hcurH = hnxtH; hnxtH = th;
  }
}
